// SharedAttention_37941741093144
// MI455X (gfx1250) — matrix operand order repaired and checked by reading, not compiled, not run on hardware
//
#include <hip/hip_runtime.h>
#include <stdint.h>

#define B_ 16
#define F_ 256
#define H_ 512
#define T_ 2048
#define KW_ 25

typedef __attribute__((ext_vector_type(16))) _Float16 v16h;
typedef __attribute__((ext_vector_type(8)))  _Float16 v8h;
typedef __attribute__((ext_vector_type(8)))  float    v8f;

static __device__ __forceinline__ v8f zero8() {
  v8f z;
#pragma unroll
  for (int i = 0; i < 8; ++i) z[i] = 0.0f;
  return z;
}
static __device__ __forceinline__ v16h mk16(v8h a, v8h b) {
  v16h r;
#pragma unroll
  for (int i = 0; i < 8; ++i) { r[i] = a[i]; r[i + 8] = b[i]; }
  return r;
}
// A-matrix 16x32 f16 fragment: lane holds row m = lane%16; K = {8g..8g+7, 16+8g..23+8g}
static __device__ __forceinline__ v16h afrag(const _Float16* p, int g) {
  return mk16(*(const v8h*)(p + 8 * g), *(const v8h*)(p + 16 + 8 * g));
}
// B-matrix 32x16 f16 fragment: lane holds col n = lane%16; K = {8g..8g+7, 16+8g..23+8g}
static __device__ __forceinline__ v16h bfrag(const _Float16* p, int g) {
  return mk16(*(const v8h*)(p + 8 * g), *(const v8h*)(p + 16 + 8 * g));
}
#define WMMA(A, Bv, C) __builtin_amdgcn_wmma_f32_16x16x32_f16(false, (A), false, (Bv), (short)0, (C), false, false)

// ---- CDNA5 async global->LDS (ASYNCcnt-tracked, no VGPR data path) ----
static __device__ __forceinline__ void async_cp16(unsigned lds_addr, const void* gp) {
  asm volatile("global_load_async_to_lds_b128 %0, %1, off"
               :: "v"(lds_addr), "v"((unsigned long long)(uintptr_t)gp)
               : "memory");
}
static __device__ __forceinline__ void wait_async0() {
#if defined(__has_builtin) && __has_builtin(__builtin_amdgcn_s_wait_asynccnt)
  __builtin_amdgcn_s_wait_asynccnt(0);
#else
  asm volatile("s_wait_asynccnt 0x0" ::: "memory");
#endif
}

// ---------------- conversion kernels ----------------

// pattern[b][f][t] (f32) -> pt[b][t][f] (f16), 32x32 LDS tile transpose
__global__ __launch_bounds__(256) void k_transpose_cvt(const float* __restrict__ src,
                                                       _Float16* __restrict__ dst) {
  __shared__ float tile[32][33];
  const int t0 = blockIdx.x * 32, f0 = blockIdx.y * 32, b = blockIdx.z;
  const int tx = threadIdx.x & 31, ty = threadIdx.x >> 5;
  const float* s = src + ((long)b * F_ + f0) * T_ + t0;
#pragma unroll
  for (int i = ty; i < 32; i += 8) tile[i][tx] = s[(long)i * T_ + tx];
  __syncthreads();
  _Float16* d = dst + ((long)b * T_ + t0) * F_ + f0;
#pragma unroll
  for (int i = ty; i < 32; i += 8) d[(long)i * F_ + tx] = (_Float16)tile[tx][i];
}

__global__ __launch_bounds__(256) void k_cvt_f16(const float* __restrict__ s,
                                                 _Float16* __restrict__ d, long n) {
  long i = (long)blockIdx.x * blockDim.x + threadIdx.x;
  long stride = (long)gridDim.x * blockDim.x;
  for (; i < n; i += stride) d[i] = (_Float16)s[i];
}

// ---------------- fused MLP: y = W2*relu(W1*x+b1)+b2 ----------------
// xT: f16 [B][T][F] (t-major). Writes f32 tile (pitch/bstride) and optional f16 transposed copy.
__global__ __launch_bounds__(256) void k_mlp(const _Float16* __restrict__ xT,
                                             const _Float16* __restrict__ W1,
                                             const float* __restrict__ b1,
                                             const _Float16* __restrict__ W2,
                                             const float* __restrict__ b2,
                                             float* __restrict__ outF, long outPitch,
                                             long outBStride,
                                             _Float16* __restrict__ outT) {
  const int b = blockIdx.y;
  const int t0 = blockIdx.x * 32;
  __shared__ __align__(16) _Float16 xB[32][264];
  __shared__ __align__(16) _Float16 hB[32][520];
  const int tid = threadIdx.x;
  {   // stage x tile: 32 t-rows x 256 f (coalesced b128 copies)
    int tt = tid >> 3, seg = (tid & 7) * 32;
    const uint4* s = (const uint4*)(xT + ((long)(b * T_ + t0 + tt)) * F_ + seg);
    uint4* d = (uint4*)&xB[tt][seg];
    d[0] = s[0]; d[1] = s[1]; d[2] = s[2]; d[3] = s[3];
  }
  __syncthreads();
  const int lane = tid & 31, wave = tid >> 5;
  const int g = lane >> 4, ln = lane & 15;

  // stage 1: h(512 x 32) : 32 mtiles x 2 ntiles, 8 tiles/wave
  for (int i = 0; i < 8; ++i) {
    int idx = wave * 8 + i, mt = idx >> 1, nt = idx & 1;
    v8f acc = zero8();
    for (int kc = 0; kc < F_; kc += 32) {
      v16h A = afrag(W1 + (long)(mt * 16 + ln) * F_ + kc, g);
      v16h Bv = bfrag(&xB[nt * 16 + ln][kc], g);
      acc = WMMA(A, Bv, acc);
    }
    v8h hv;
#pragma unroll
    for (int r = 0; r < 8; ++r) {
      float v = acc[r] + b1[mt * 16 + 8 * g + r];
      hv[r] = (_Float16)fmaxf(v, 0.0f);
    }
    *(v8h*)(&hB[nt * 16 + ln][mt * 16 + 8 * g]) = hv;  // contiguous in r
  }
  __syncthreads();

  // stage 2: y(256 x 32) : 16 mtiles x 2 ntiles, 4 tiles/wave
  for (int i = 0; i < 4; ++i) {
    int idx = wave * 4 + i, mt = idx >> 1, nt = idx & 1;
    v8f acc = zero8();
    for (int kc = 0; kc < H_; kc += 32) {
      v16h A = afrag(W2 + (long)(mt * 16 + ln) * H_ + kc, g);
      v16h Bv = bfrag(&hB[nt * 16 + ln][kc], g);
      acc = WMMA(A, Bv, acc);
    }
    int tcol = t0 + nt * 16 + ln;
    v8h ov;
#pragma unroll
    for (int r = 0; r < 8; ++r) {
      int f = mt * 16 + 8 * g + r;
      float v = acc[r] + b2[f];
      outF[(long)b * outBStride + (long)f * outPitch + tcol] = v;
      ov[r] = (_Float16)v;
    }
    if (outT)
      *(v8h*)(outT + ((long)(b * T_ + tcol)) * F_ + mt * 16 + 8 * g) = ov;
  }
}

// ---------------- flash attention with double-exp softmax ----------------
// qT,kT: f16 [B][T][F]; vN: f16 [B][F][T]; aT out: f16 [B][T][F]
// One wave owns a 16-t tile; v chunk double-buffered in LDS via async DMA;
// k B-fragments come straight from global (identical across the 8 waves -> L0).
__global__ __launch_bounds__(256) void k_attn(const _Float16* __restrict__ qT,
                                              const _Float16* __restrict__ kT,
                                              const _Float16* __restrict__ vN,
                                              _Float16* __restrict__ aT) {
  const int b = blockIdx.y;
  const int tid = threadIdx.x;
  const int lane = tid & 31, wave = tid >> 5;
  const int g = lane >> 4, ln = lane & 15;
  const int t0 = blockIdx.x * 128 + wave * 16;

  __shared__ __align__(16) _Float16 vB[2][256][40];  // v chunk, [d][s], double-buffered
  __shared__ __align__(16) _Float16 pB[8][16][32];   // P per wave, [t][s]
  __shared__ float cSw[8][16];
  __shared__ float sSw[8][16];

  // async prefetch of a 256d x 32s v chunk: 4 x b128 per thread (row d = tid)
  const _Float16* vRow = vN + ((long)(b * F_ + tid)) * T_;
  auto stageV = [&](int buf, int s0n) {
    const _Float16* src = vRow + s0n;
    unsigned dst = (unsigned)(uintptr_t)&vB[buf][tid][0];
    async_cp16(dst, src);
    async_cp16(dst + 16, src + 8);
    async_cp16(dst + 32, src + 16);
    async_cp16(dst + 48, src + 24);
  };

  // preload q A-fragments (loop-invariant)
  v16h aq[8];
  const _Float16* qrow = qT + ((long)(b * T_ + t0 + ln)) * F_;
#pragma unroll
  for (int kcI = 0; kcI < 8; ++kcI) aq[kcI] = afrag(qrow + kcI * 32, g);

  v8f o_[16];
#pragma unroll
  for (int i = 0; i < 16; ++i) o_[i] = zero8();
  float lmax[8], rsum[8];
#pragma unroll
  for (int r = 0; r < 8; ++r) { lmax[r] = 0.0f; rsum[r] = 0.0f; }

  const float diagScale = 1.0f - 0.02209708691207961f;  // 1 - 1/sqrt(2048)

  stageV(0, 0);  // prologue

  for (int s0 = 0, it = 0; s0 < T_; s0 += 32, ++it) {
    const int cur = it & 1;
    wait_async0();     // own async copies done
    __syncthreads();   // => every wave's copies done; prev buffer fully consumed
    if (s0 + 32 < T_) stageV(cur ^ 1, s0 + 32);

    // S tiles (two 16x16, s-halves), K = 256; k B-frags direct from global
    const _Float16* k0 = kT + ((long)(b * T_ + s0 + ln)) * F_;
    const _Float16* k1 = k0 + 16 * F_;
    if (s0 + 32 < T_) {  // hint next k chunk into cache
      __builtin_prefetch(k0 + 32 * F_, 0, 3);
      __builtin_prefetch(k1 + 32 * F_, 0, 3);
    }
    v8f st0 = zero8(), st1 = zero8();
#pragma unroll
    for (int kcI = 0; kcI < 8; ++kcI) {
      v16h B0 = bfrag(k0 + kcI * 32, g);
      v16h B1 = bfrag(k1 + kcI * 32, g);
      st0 = WMMA(aq[kcI], B0, st0);
      st1 = WMMA(aq[kcI], B1, st1);
    }
    // diagonal: logits = exp(S - S/sqrt(T)) on t==s (chunk overlaps this t-tile only rarely)
    if (t0 + 15 >= s0 && t0 <= s0 + 31) {
#pragma unroll
      for (int r = 0; r < 8; ++r) {
        int tg = t0 + r + 8 * g;
        if (s0 + ln == tg) st0[r] *= diagScale;
        if (s0 + 16 + ln == tg) st1[r] *= diagScale;
      }
    }
    // chunk row-max of S over the 32 s values
    float cm[8];
#pragma unroll
    for (int r = 0; r < 8; ++r) cm[r] = fmaxf(st0[r], st1[r]);
#pragma unroll
    for (int m = 1; m < 16; m <<= 1)
#pragma unroll
      for (int r = 0; r < 8; ++r) cm[r] = fmaxf(cm[r], __shfl_xor(cm[r], m, 32));

    // online softmax in logit (L = exp(S)) space
    float cf[8], cs[8];
    v8h p0, p1;
#pragma unroll
    for (int r = 0; r < 8; ++r) {
      float lnew = fmaxf(lmax[r], expf(cm[r]));
      cf[r] = expf(lmax[r] - lnew);
      lmax[r] = lnew;
      float e0 = expf(expf(st0[r]) - lnew);
      float e1 = expf(expf(st1[r]) - lnew);
      cs[r] = e0 + e1;
      p0[r] = (_Float16)e0;
      p1[r] = (_Float16)e1;
    }
#pragma unroll
    for (int m = 1; m < 16; m <<= 1)
#pragma unroll
      for (int r = 0; r < 8; ++r) cs[r] += __shfl_xor(cs[r], m, 32);
#pragma unroll
    for (int r = 0; r < 8; ++r) rsum[r] = rsum[r] * cf[r] + cs[r];

    // P -> LDS as [t][s]; rescale factor -> per-lane-by-t via LDS swap
#pragma unroll
    for (int r = 0; r < 8; ++r) {
      pB[wave][r + 8 * g][ln] = p0[r];
      pB[wave][r + 8 * g][16 + ln] = p1[r];
    }
    if (ln == 0) {
#pragma unroll
      for (int r = 0; r < 8; ++r) cSw[wave][r + 8 * g] = cf[r];
    }
    float cl = cSw[wave][ln];  // same-wave LDS ops are in order

    // O = O*cl + v * P^T  (16 d-tiles, K = 32); skip rescale when max unchanged
    bool needScale = !__all(cl == 1.0f);
#pragma unroll
    for (int mt = 0; mt < 16; ++mt) {
      v16h A = afrag(&vB[cur][mt * 16 + ln][0], g);
      v16h Bv = bfrag(&pB[wave][ln][0], g);
      if (needScale) o_[mt] *= cl;
      o_[mt] = WMMA(A, Bv, o_[mt]);
    }
  }

  if (ln == 0) {
#pragma unroll
    for (int r = 0; r < 8; ++r) sSw[wave][r + 8 * g] = rsum[r];
  }
  float inv = 1.0f / sSw[wave][ln];
#pragma unroll
  for (int mt = 0; mt < 16; ++mt) {
    v8h ov;
#pragma unroll
    for (int r = 0; r < 8; ++r) ov[r] = (_Float16)(o_[mt][r] * inv);
    *(v8h*)(aT + ((long)(b * T_ + t0 + ln)) * F_ + mt * 16 + 8 * g) = ov;
  }
}

// ---------------- ex path: uniform attention (eye(1) broadcast) + tiny MLP ----------------
__global__ __launch_bounds__(256) void k_ex(const float* __restrict__ value,
                                            const float* __restrict__ W1,
                                            const float* __restrict__ b1,
                                            const float* __restrict__ W2,
                                            const float* __restrict__ b2,
                                            float* __restrict__ rep) {
  const int b = blockIdx.x, tid = threadIdx.x;
  __shared__ float mF[F_];
  __shared__ float hH[H_];
  const float* vrow = value + ((long)b * F_ + tid) * T_;
  float s = 0.0f;
  for (int i = KW_; i < T_ - 1; ++i) s += vrow[i];
  mF[tid] = s / (float)(T_ - 1 - KW_);
  __syncthreads();
  for (int j = tid; j < H_; j += 256) {
    const float* w = W1 + (long)j * F_;
    float a = b1[j];
    for (int f = 0; f < F_; ++f) a += w[f] * mF[f];
    hH[j] = fmaxf(a, 0.0f);
  }
  __syncthreads();
  {
    const float* w = W2 + (long)tid * H_;
    float a = b2[tid];
    for (int j = 0; j < H_; ++j) a += w[j] * hH[j];
    rep[((long)b * F_ + tid) * (T_ + 1) + T_] = a;
  }
}

// ---------------- launcher ----------------
extern "C" void kernel_launch(void* const* d_in, const int* in_sizes, int n_in,
                              void* d_out, int out_size, void* d_ws, size_t ws_size,
                              hipStream_t stream) {
  const float* pattern = (const float*)d_in[0];
  const float* value = (const float*)d_in[1];
  const float* W1q = (const float*)d_in[2];
  const float* b1q = (const float*)d_in[3];
  const float* W2q = (const float*)d_in[4];
  const float* b2q = (const float*)d_in[5];
  const float* W1k = (const float*)d_in[6];
  const float* b1k = (const float*)d_in[7];
  const float* W2k = (const float*)d_in[8];
  const float* b2k = (const float*)d_in[9];
  const float* W1o = (const float*)d_in[10];
  const float* b1o = (const float*)d_in[11];
  const float* W2o = (const float*)d_in[12];
  const float* b2o = (const float*)d_in[13];

  float* rep = (float*)d_out;                       // [B][F][T+1]
  float* qout = rep + (size_t)B_ * F_ * (T_ + 1);   // [B][F][T]
  float* kout = qout + (size_t)B_ * F_ * T_;        // [B][F][T]

  const size_t NBFT = (size_t)B_ * F_ * T_;
  const size_t NW = (size_t)H_ * F_;
  _Float16* pt = (_Float16*)d_ws;      // pattern^T f16 [B][T][F]
  _Float16* vh = pt + NBFT;            // value f16   [B][F][T]
  _Float16* qt = vh + NBFT;            // q^T f16
  _Float16* kt = qt + NBFT;            // k^T f16
  _Float16* at = kt + NBFT;            // attn out^T f16
  _Float16* w1qh = at + NBFT;
  _Float16* w2qh = w1qh + NW;
  _Float16* w1kh = w2qh + NW;
  _Float16* w2kh = w1kh + NW;
  _Float16* w1oh = w2kh + NW;
  _Float16* w2oh = w1oh + NW;

  k_transpose_cvt<<<dim3(T_ / 32, F_ / 32, B_), 256, 0, stream>>>(pattern, pt);
  k_cvt_f16<<<dim3(2048), 256, 0, stream>>>(value, vh, (long)NBFT);
  k_cvt_f16<<<dim3(64), 256, 0, stream>>>(W1q, w1qh, (long)NW);
  k_cvt_f16<<<dim3(64), 256, 0, stream>>>(W2q, w2qh, (long)NW);
  k_cvt_f16<<<dim3(64), 256, 0, stream>>>(W1k, w1kh, (long)NW);
  k_cvt_f16<<<dim3(64), 256, 0, stream>>>(W2k, w2kh, (long)NW);
  k_cvt_f16<<<dim3(64), 256, 0, stream>>>(W1o, w1oh, (long)NW);
  k_cvt_f16<<<dim3(64), 256, 0, stream>>>(W2o, w2oh, (long)NW);

  k_mlp<<<dim3(T_ / 32, B_), 256, 0, stream>>>(pt, w1qh, b1q, w2qh, b2q, qout,
                                               (long)T_, (long)F_ * T_, qt);
  k_mlp<<<dim3(T_ / 32, B_), 256, 0, stream>>>(pt, w1kh, b1k, w2kh, b2k, kout,
                                               (long)T_, (long)F_ * T_, kt);
  k_attn<<<dim3(T_ / 128, B_), 256, 0, stream>>>(qt, kt, vh, at);
  k_mlp<<<dim3(T_ / 32, B_), 256, 0, stream>>>(at, w1oh, b1o, w2oh, b2o, rep,
                                               (long)(T_ + 1), (long)F_ * (T_ + 1),
                                               nullptr);
  k_ex<<<dim3(B_), 256, 0, stream>>>(value, W1o, b1o, W2o, b2o, rep);
}
